// MultiBoxLoss_86388972191984
// MI455X (gfx1250) — compile-verified, matrix operand order checked by reading, not run on hardware
//
#include <hip/hip_runtime.h>
#include <math.h>

#define B_  64
#define P_  8732
#define C_  81
#define O_  32
#define THRESH   0.5f
#define NPRATIO  3
#define ALPHA_   1.0f
#define SORT_N   16384   // next pow2 >= P_

typedef int      v8i __attribute__((ext_vector_type(8)));
typedef unsigned v4u __attribute__((ext_vector_type(4)));
typedef unsigned v8u __attribute__((ext_vector_type(8)));

__device__ __forceinline__ float wave_max32(float v) {
    #pragma unroll
    for (int m = 16; m > 0; m >>= 1) v = fmaxf(v, __shfl_xor(v, m, 32));
    return v;
}
__device__ __forceinline__ float wave_sum32(float v) {
    #pragma unroll
    for (int m = 16; m > 0; m >>= 1) v += __shfl_xor(v, m, 32);
    return v;
}
__device__ __forceinline__ int wave_isum32(int v) {
    #pragma unroll
    for (int m = 16; m > 0; m >>= 1) v += __shfl_xor(v, m, 32);
    return v;
}

// ---------------------------------------------------------------- init
__global__ void mb_init_kernel(float* __restrict__ accum) {
    if (threadIdx.x < 8) accum[threadIdx.x] = 0.0f;
}

// ---------------------------------------------------------------- match
__global__ __launch_bounds__(256)
void mb_match_kernel(const float* __restrict__ boxes,    // [B,O,4] xyxy
                     const int*   __restrict__ labels,   // [B,O]
                     const float* __restrict__ priors,   // [P,4] cxcywh
                     float* __restrict__ ovfp,           // [B,P]
                     int*   __restrict__ objfp,          // [B,P]
                     float* __restrict__ true_locs,      // [B,P,4]
                     int*   __restrict__ true_cls)       // [B,P]
{
    const int b   = blockIdx.x;
    const int tid = threadIdx.x;

    __shared__ float bx0[O_], by0[O_], bx1[O_], by1[O_], barea[O_];
    __shared__ int   blab[O_];
    __shared__ unsigned long long bestkey[O_];

    if (tid < O_) {
        float x0 = boxes[(b*O_ + tid)*4 + 0];
        float y0 = boxes[(b*O_ + tid)*4 + 1];
        float x1 = boxes[(b*O_ + tid)*4 + 2];
        float y1 = boxes[(b*O_ + tid)*4 + 3];
        bx0[tid] = x0; by0[tid] = y0; bx1[tid] = x1; by1[tid] = y1;
        barea[tid] = (x1 - x0) * (y1 - y0);
        blab[tid]  = labels[b*O_ + tid];
        bestkey[tid] = 0ULL;
    }
    __syncthreads();

    // Phase A: per-prior argmax over objects; feed per-object argmax keys.
    for (int p = tid; p < P_; p += blockDim.x) {
        float cx = priors[p*4+0], cy = priors[p*4+1];
        float pw = priors[p*4+2], ph = priors[p*4+3];
        float px0 = cx - 0.5f*pw, py0 = cy - 0.5f*ph;
        float px1 = cx + 0.5f*pw, py1 = cy + 0.5f*ph;
        float parea = pw * ph;

        float best = -1.0f; int besto = 0;
        #pragma unroll 4
        for (int o = 0; o < O_; ++o) {
            float iw = fminf(bx1[o], px1) - fmaxf(bx0[o], px0);
            float ih = fminf(by1[o], py1) - fmaxf(by0[o], py0);
            iw = fmaxf(iw, 0.0f); ih = fmaxf(ih, 0.0f);
            float inter = iw * ih;
            float iou = inter / (barea[o] + parea - inter);
            if (iou > best) { best = iou; besto = o; }
            unsigned long long key =
                ((unsigned long long)__float_as_uint(iou) << 32) |
                (unsigned long long)(0xFFFFFFFFu - (unsigned)p);
            if (key > bestkey[o]) atomicMax(&bestkey[o], key);
        }
        ovfp [b*P_ + p] = best;
        objfp[b*P_ + p] = besto;
    }
    __syncthreads();

    // Phase B: bijective override + label + box encoding.
    for (int p = tid; p < P_; p += blockDim.x) {
        float best  = ovfp [b*P_ + p];
        int   besto = objfp[b*P_ + p];
        unsigned pk = 0xFFFFFFFFu - (unsigned)p;
        #pragma unroll 4
        for (int o = 0; o < O_; ++o) {
            if ((unsigned)(bestkey[o] & 0xFFFFFFFFULL) == pk) { besto = o; best = 1.0f; }
        }
        int lab = (best < THRESH) ? 0 : blab[besto];

        float x0 = bx0[besto], y0 = by0[besto], x1 = bx1[besto], y1 = by1[besto];
        float gcx = 0.5f*(x0 + x1), gcy = 0.5f*(y0 + y1);
        float gw  = x1 - x0,        gh  = y1 - y0;
        float pcx = priors[p*4+0], pcy = priors[p*4+1];
        float pw  = priors[p*4+2], ph  = priors[p*4+3];

        long long base = ((long long)b*P_ + p) * 4;
        true_locs[base+0] = (gcx - pcx) / (pw * 0.1f);
        true_locs[base+1] = (gcy - pcy) / (ph * 0.1f);
        true_locs[base+2] = logf(gw / pw) * 5.0f;
        true_locs[base+3] = logf(gh / ph) * 5.0f;
        true_cls[b*P_ + p] = lab;
    }
}

// ---------------------------------------------------------------- count (WMMA)
// One wave per batch. 8 unguarded WMMA iterations (b128 loads, 1024 mask
// bytes each -> exact IU8 WMMA row sums), then a scalar tail of 540.
__global__ __launch_bounds__(32)
void mb_count_pos_kernel(const int* __restrict__ true_cls, int* __restrict__ n_pos)
{
    const int b    = blockIdx.x;
    const int lane = threadIdx.x;

    v8i acc  = {0,0,0,0,0,0,0,0};
    v8i ones;
    #pragma unroll
    for (int k = 0; k < 8; ++k) ones[k] = 0x01010101;

    const int4* src = (const int4*)(true_cls + (long long)b * P_);
    #pragma unroll 1
    for (int t = 0; t < 8; ++t) {                 // 8*1024 = 8192 labels
        v8i a;
        #pragma unroll
        for (int d = 0; d < 8; ++d) {
            int4 q = src[t*256 + lane*8 + d];     // 32 consecutive ints / lane
            unsigned w = (q.x != 0 ? 1u : 0u)
                       | (q.y != 0 ? 0x100u : 0u)
                       | (q.z != 0 ? 0x10000u : 0u)
                       | (q.w != 0 ? 0x1000000u : 0u);
            a[d] = (int)w;
        }
        acc = __builtin_amdgcn_wmma_i32_16x16x64_iu8(
                  false, a, false, ones, acc, false, false);
    }
    int s = 0;
    #pragma unroll
    for (int k = 0; k < 8; ++k) s += acc[k];
    int t0  = __shfl(s, 0, 32);
    int t16 = __shfl(s, 16, 32);

    int rem = 0;                                   // tail: 8192..8731
    for (int idx = 8192 + lane; idx < P_; idx += 32)
        rem += (true_cls[(long long)b*P_ + idx] != 0) ? 1 : 0;
    rem = wave_isum32(rem);

    if (lane == 0) n_pos[b] = t0 + t16 + rem;
}

// ---------------------------------------------------------------- CE
// One wave32 per prior. The 81 logits are staged into LDS with per-lane
// async global->LDS copies (ASYNCcnt), completed with s_wait_asynccnt,
// then reduced with wave32 shuffles.
__global__ __launch_bounds__(256)
void mb_ce_kernel(const float* __restrict__ scores,      // [B,P,C]
                  const float* __restrict__ pred_locs,   // [B,P,4]
                  const float* __restrict__ true_locs,   // [B,P,4]
                  const int*   __restrict__ true_cls,    // [B,P]
                  float* __restrict__ ce,                // [B,P]
                  float* __restrict__ accum)             // [0]=pos_ce [1]=loc
{
    __shared__ float sbuf[8 * 96];
    __shared__ float blk[2];
    const int tid  = threadIdx.x;
    const int wave = tid >> 5;
    const int lane = tid & 31;
    if (tid < 2) blk[tid] = 0.0f;
    __syncthreads();

    long long i = (long long)blockIdx.x * 8 + wave;
    float wpos = 0.0f, wloc = 0.0f;
    if (i < (long long)B_ * P_) {
        const float* gp = scores + i * C_;
        unsigned sbase = (unsigned)(unsigned long long)(&sbuf[wave * 96]);

        unsigned l0 = sbase + (unsigned)lane * 4u;
        unsigned l1 = sbase + (32u + (unsigned)lane) * 4u;
        unsigned l2 = sbase + (64u + (unsigned)lane) * 4u;
        unsigned long long a0 = (unsigned long long)(gp + lane);
        unsigned long long a1 = (unsigned long long)(gp + 32 + lane);
        int col2 = (lane < C_ - 64) ? (64 + lane) : lane;      // clamp in-bounds
        unsigned long long a2 = (unsigned long long)(gp + col2);

        asm volatile("global_load_async_to_lds_b32 %0, %1, off"
                     :: "v"(l0), "v"(a0) : "memory");
        asm volatile("global_load_async_to_lds_b32 %0, %1, off"
                     :: "v"(l1), "v"(a1) : "memory");
        asm volatile("global_load_async_to_lds_b32 %0, %1, off"
                     :: "v"(l2), "v"(a2) : "memory");
        asm volatile("s_wait_asynccnt 0x0" ::: "memory");

        float s0 = sbuf[wave*96 + lane];
        float s1 = sbuf[wave*96 + 32 + lane];
        float s2 = (lane < C_ - 64) ? sbuf[wave*96 + 64 + lane] : -INFINITY;

        float m  = wave_max32(fmaxf(fmaxf(s0, s1), s2));
        float e  = expf(s0 - m) + expf(s1 - m) +
                   ((lane < C_ - 64) ? expf(s2 - m) : 0.0f);
        e = wave_sum32(e);
        float lse = m + logf(e);
        int lab = true_cls[i];
        if (lane == 0) {
            float cv = lse - gp[lab];
            ce[i] = cv;
            if (lab != 0) {
                wpos = cv;
                float4 pl = *(const float4*)(pred_locs + i*4);
                float4 tl = *(const float4*)(true_locs + i*4);
                wloc = fabsf(pl.x - tl.x) + fabsf(pl.y - tl.y)
                     + fabsf(pl.z - tl.z) + fabsf(pl.w - tl.w);
            }
        }
    }
    if (lane == 0 && (wpos != 0.0f || wloc != 0.0f)) {
        atomicAdd(&blk[0], wpos);
        atomicAdd(&blk[1], wloc);
    }
    __syncthreads();
    if (tid == 0 && blk[0] != 0.0f) atomicAdd(&accum[0], blk[0]);
    if (tid == 1 && blk[1] != 0.0f) atomicAdd(&accum[1], blk[1]);
}

// ---------------------------------------------------------------- hard negatives
// One block per batch. The batch's ce[] row (34.9 KB contiguous) is DMA'd
// into the LDS sort array by the Tensor Data Mover (1D tile descriptor,
// issued by wave 0, completed with s_wait_tensorcnt), then positives are
// zeroed, bitonic-sorted descending, and the top 3*n_pos are summed.
__global__ __launch_bounds__(1024)
void mb_hardneg_kernel(const float* __restrict__ ce,
                       const int*   __restrict__ true_cls,
                       const int*   __restrict__ n_pos,
                       float* __restrict__ accum)        // [2]=hard_neg
{
    __shared__ float arr[SORT_N];
    __shared__ float ssum;
    const int b   = blockIdx.x;
    const int tid = threadIdx.x;
    const int nth = blockDim.x;

    if (tid < 32) {
        if (tid == 0) ssum = 0.0f;
        unsigned long long ga = (unsigned long long)(ce + (long long)b * P_);
        unsigned lds = (unsigned)(unsigned long long)(&arr[0]);

        v4u g0;
        g0[0] = 1u;                                        // count=1 (valid D#)
        g0[1] = lds;                                       // lds_addr
        g0[2] = (unsigned)(ga & 0xFFFFFFFFu);              // global_addr[31:0]
        g0[3] = (unsigned)((ga >> 32) & 0x1FFFFFFu)        // global_addr[56:32]
              | (2u << 30);                                // type=2 (image)

        v8u g1;
        g1[0] = (2u << 16);                   // data_size=4B, wg_mask=0
        g1[1] = ((unsigned)P_ & 0xFFFFu) << 16;            // tensor_dim0[15:0]
        g1[2] = (((unsigned)P_ >> 16) & 0xFFFFu)           // tensor_dim0[31:16]
              | (1u << 16);                                // tensor_dim1 = 1
        g1[3] = ((unsigned)P_ & 0xFFFFu) << 16;            // tile_dim0 = P_
        g1[4] = 0u;                                        // tile_dim1/2 unused
        g1[5] = (unsigned)P_;                              // tensor_dim0_stride
        g1[6] = 0u;
        g1[7] = 0u;

        asm volatile("tensor_load_to_lds %0, %1"
                     :: "s"(g0), "s"(g1) : "memory");
        __builtin_amdgcn_s_wait_tensorcnt(0);
    }
    __syncthreads();

    // zero positives, zero pad (CE >= 0, so 0-padding is exact)
    for (int i = tid; i < SORT_N; i += nth) {
        if (i < P_) {
            if (true_cls[(long long)b*P_ + i] != 0) arr[i] = 0.0f;
        } else {
            arr[i] = 0.0f;
        }
    }
    __syncthreads();

    // bitonic sort, descending
    for (unsigned k = 2; k <= SORT_N; k <<= 1) {
        for (unsigned j = k >> 1; j > 0; j >>= 1) {
            for (unsigned i = tid; i < SORT_N; i += nth) {
                unsigned ixj = i ^ j;
                if (ixj > i) {
                    float a = arr[i], c = arr[ixj];
                    bool down = ((i & k) == 0);
                    if (down ? (a < c) : (a > c)) { arr[i] = c; arr[ixj] = a; }
                }
            }
            __syncthreads();
        }
    }

    int K = NPRATIO * n_pos[b];
    if (K > SORT_N) K = SORT_N;
    float s = 0.0f;
    for (int i = tid; i < K; i += nth) s += arr[i];
    s = wave_sum32(s);
    if ((tid & 31) == 0 && s != 0.0f) atomicAdd(&ssum, s);
    __syncthreads();
    if (tid == 0) atomicAdd(&accum[2], ssum);
}

// ---------------------------------------------------------------- final
__global__ void mb_final_kernel(const float* __restrict__ accum,
                                const int*   __restrict__ n_pos,
                                float* __restrict__ out)
{
    if (threadIdx.x == 0 && blockIdx.x == 0) {
        int npt = 0;
        for (int b = 0; b < B_; ++b) npt += n_pos[b];
        float n = (float)npt;
        float loc_loss  = accum[1] / (n * 4.0f);
        float conf_loss = (accum[2] + accum[0]) / n;
        out[0] = conf_loss + ALPHA_ * loc_loss;
    }
}

// ---------------------------------------------------------------- launch
extern "C" void kernel_launch(void* const* d_in, const int* in_sizes, int n_in,
                              void* d_out, int out_size, void* d_ws, size_t ws_size,
                              hipStream_t stream) {
    const float* pred_locs   = (const float*)d_in[0];  // [B,P,4]
    const float* pred_scores = (const float*)d_in[1];  // [B,P,C]
    const float* boxes       = (const float*)d_in[2];  // [B,O,4]
    const int*   labels      = (const int*)  d_in[3];  // [B,O]
    const float* priors      = (const float*)d_in[4];  // [P,4]
    float* out = (float*)d_out;

    const long long NP = (long long)B_ * P_;
    char* w = (char*)d_ws;
    float* true_locs = (float*)w;                 w += sizeof(float) * NP * 4;
    int*   true_cls  = (int*)w;                   w += sizeof(int)   * NP;
    float* ce        = (float*)w;                 w += sizeof(float) * NP;
    float* ovfp      = (float*)w;                 w += sizeof(float) * NP;
    int*   objfp     = (int*)w;                   w += sizeof(int)   * NP;
    int*   n_pos     = (int*)w;                   w += sizeof(int)   * 256;
    float* accum     = (float*)w;                 // 8 floats

    mb_init_kernel<<<1, 32, 0, stream>>>(accum);
    mb_match_kernel<<<B_, 256, 0, stream>>>(boxes, labels, priors,
                                            ovfp, objfp, true_locs, true_cls);
    mb_count_pos_kernel<<<B_, 32, 0, stream>>>(true_cls, n_pos);
    int nblocks = (int)((NP + 7) / 8);
    mb_ce_kernel<<<nblocks, 256, 0, stream>>>(pred_scores, pred_locs, true_locs,
                                              true_cls, ce, accum);
    mb_hardneg_kernel<<<B_, 1024, 0, stream>>>(ce, true_cls, n_pos, accum);
    mb_final_kernel<<<1, 1, 0, stream>>>(accum, n_pos, out);
}
